// DoubleWindowAttention_89876485636823
// MI455X (gfx1250) — matrix operand order repaired and checked by reading, not compiled, not run on hardware
//
#include <hip/hip_runtime.h>
#include <hip/hip_bf16.h>

// ---------------------------------------------------------------------------
// CDNA5 (gfx1250) wave32 WMMA types/helpers
// ---------------------------------------------------------------------------
typedef __attribute__((ext_vector_type(16))) __bf16 v16bf;
typedef __attribute__((ext_vector_type(8)))  __bf16 v8bf;
typedef __attribute__((ext_vector_type(8)))  float  v8f;
typedef __attribute__((ext_vector_type(4)))  unsigned int u32x4;
typedef __attribute__((ext_vector_type(8)))  unsigned int u32x8;

union FragBF { v16bf v; v8bf h[2]; };

#define WMMA_BF16(A, B, C) \
    __builtin_amdgcn_wmma_f32_16x16x32_bf16(false, (A), false, (B), (short)0, (C), false, false)

// A fragment (16x32 bf16, MxK): lane m holds K = {ka..ka+7, ka+16..ka+23}, ka = (lane<16)?0:8
__device__ __forceinline__ v16bf load_fragA(const __bf16* rowptr, int lane) {
    FragBF f;
    int ka = (lane & 16) ? 8 : 0;
    f.h[0] = *(const v8bf*)(rowptr + ka);
    f.h[1] = *(const v8bf*)(rowptr + ka + 16);
    return f.v;
}
// B fragment (32x16 bf16, KxN) loaded from B^T rows: lane n holds K = {kb..kb+7, kb+16..kb+23}, kb=(lane<16)?0:8
__device__ __forceinline__ v16bf load_fragB(const __bf16* colptr, int lane) {
    FragBF f;
    int kb = (lane & 16) ? 8 : 0;
    f.h[0] = *(const v8bf*)(colptr + kb);
    f.h[1] = *(const v8bf*)(colptr + kb + 16);
    return f.v;
}

// ds_swizzle_b32 XOR butterfly within 16-lane halves (no per-lane address math)
template <int XORMASK>
__device__ __forceinline__ float swz_xor(float x) {
    // group-of-32 mode: offset[15]=0, and_mask=0x1f, or_mask=0, xor_mask=XORMASK
    return __int_as_float(
        __builtin_amdgcn_ds_swizzle(__float_as_int(x), (XORMASK << 10) | 0x1f));
}

// ---------------------------------------------------------------------------
// Tensor Data Mover: 2D tile (global -> LDS), D# built in SGPR tuples.
//   d0flags carries data_size + LDS padding config (group1 dword0).
//   tensor dims == tile dims (tile start passed as global_addr, no OOB).
// ---------------------------------------------------------------------------
__device__ __forceinline__ void tdm_load_2d(unsigned int lds_off, const void* gptr,
                                            unsigned int tile_d0, unsigned int tile_d1,
                                            unsigned int stride0, unsigned int d0flags) {
    unsigned long long ga = (unsigned long long)(uintptr_t)gptr;
    u32x4 g0;
    g0[0] = 1u;                                    // count=1, user mode
    g0[1] = lds_off;                               // lds_addr (bytes)
    g0[2] = (unsigned int)ga;                      // global_addr[31:0]
    g0[3] = (unsigned int)((ga >> 32) & 0x01FFFFFFu) | 0x80000000u;  // addr[56:32] | type=2
    u32x8 g1;
    g1[0] = d0flags;                               // wg_mask=0 | data_size | pad cfg
    g1[1] = (tile_d0 & 0xFFFFu) << 16;             // tensor_dim0[15:0]
    g1[2] = ((tile_d0 >> 16) & 0xFFFFu) | ((tile_d1 & 0xFFFFu) << 16);   // t_dim0 hi | t_dim1 lo
    g1[3] = ((tile_d1 >> 16) & 0xFFFFu) | ((tile_d0 & 0xFFFFu) << 16);   // t_dim1 hi | tile_dim0
    g1[4] = (tile_d1 & 0xFFFFu);                   // tile_dim1 | tile_dim2=0
    g1[5] = stride0;                               // tensor_dim0_stride[31:0] (elements)
    g1[6] = 0u;
    g1[7] = 0u;
    asm volatile("tensor_load_to_lds %0, %1" :: "s"(g0), "s"(g1) : "memory");
}

// bf16 tiles: data_size=2B (code 1), pad every 16 DWORDs (64B row) by 4 DWORDs (8 elems)
#define TDM_BF16_PAD40 ((1u << 16) | (1u << 20) | (3u << 22) | (3u << 25))
// f32 tiles: data_size=4B (code 2), no padding
#define TDM_F32_FLAT   (2u << 16)

// ---------------------------------------------------------------------------
// Problem constants
// ---------------------------------------------------------------------------
#define BB     32
#define CC     256
#define HW     4096      // H*W
#define WS_    8
#define SHIFT_ 4
#define HEADS_ 8
#define DK_    32
#define LL     64        // WS*WS
#define OQKV   768       // q(256)+k(256)+v(256) output channels

// ---------------------------------------------------------------------------
// Kernel 0: weight conversion + relative-position-bias table
// ---------------------------------------------------------------------------
__global__ void __launch_bounds__(256)
prep_kernel(const float* __restrict__ wq,   const float* __restrict__ bq,
            const float* __restrict__ wkv,  const float* __restrict__ bkv,
            const float* __restrict__ wproj,const float* __restrict__ rpb,
            __bf16* __restrict__ Wall, __bf16* __restrict__ Wproj,
            float* __restrict__ ball, float* __restrict__ biasTab) {
    int tid = blockIdx.x * blockDim.x + threadIdx.x;
    int nth = gridDim.x * blockDim.x;
    for (int i = tid; i < OQKV * CC; i += nth) {
        int o = i >> 8, c = i & 255;
        float w = (o < CC) ? wq[o * CC + c] : wkv[(o - CC) * CC + c];
        Wall[i] = (__bf16)w;
    }
    for (int i = tid; i < CC * CC; i += nth) Wproj[i] = (__bf16)wproj[i];
    for (int i = tid; i < OQKV; i += nth)    ball[i]  = (i < CC) ? bq[i] : bkv[i - CC];
    for (int i = tid; i < HEADS_ * LL * LL; i += nth) {
        int hd = i >> 12; int ts = i & 4095;
        int t = ts >> 6, s = ts & 63;
        int ti = t >> 3, tj = t & 7, si = s >> 3, sj = s & 7;
        int rel = (ti - si + WS_ - 1) * (2 * WS_ - 1) + (tj - sj + WS_ - 1);
        biasTab[i] = rpb[rel * HEADS_ + hd];
    }
}

// ---------------------------------------------------------------------------
// GEMM: Y[b][o][p] = sum_c W[o][c] * X[b][c][p] + bias[o]
//   block: 256 threads (8 waves), tile M=128 x N=128, K staged in chunks of 32.
//   A (weights) tiles: TDM double-buffered (tensor_load_to_lds, TENSORcnt pipeline).
//   B (activations): manual stage w/ f32->bf16 conversion into padded B^T layout.
// ---------------------------------------------------------------------------
template <typename TIN, typename TOUT, int MTOT>
__global__ void __launch_bounds__(256)
gemm_kernel(const TIN* __restrict__ X, const __bf16* __restrict__ W,
            const float* __restrict__ bias, TOUT* __restrict__ Y) {
    const int MT = MTOT / 128;
    int bi  = blockIdx.x;
    int b   = bi / (MT * 32);
    int rem = bi % (MT * 32);
    int m0  = (rem / 32) * 128;
    int p0  = (rem % 32) * 128;

    __shared__ __bf16 alds[2][128 * 40];  // A tiles [m][k] (TDM pads 32->40 stride)
    __shared__ __bf16 blds[128 * 40];     // B^T tile [p][k]

    int tid  = threadIdx.x;
    int lane = tid & 31;
    int w    = tid >> 5;
    int wm   = w >> 2;
    int wn   = w & 3;
    int lm   = lane & 15;

    v8f acc[4][2];
    v8f z = {0.f, 0.f, 0.f, 0.f, 0.f, 0.f, 0.f, 0.f};
#pragma unroll
    for (int mi = 0; mi < 4; ++mi)
#pragma unroll
        for (int ni = 0; ni < 2; ++ni) acc[mi][ni] = z;

    const TIN* xb = X + (size_t)b * CC * HW;

    // prime stage 0: TDM the first 128x32 weight tile
    if (tid < 32) {
        tdm_load_2d((unsigned int)(uintptr_t)&alds[0][0],
                    W + (size_t)m0 * CC, 32u, 128u, (unsigned int)CC, TDM_BF16_PAD40);
    }

    for (int ks = 0; ks < 8; ++ks) {
        int k0 = ks * 32;
        // TDM next weight tile into the other buffer (in-order per wave)
        if (ks < 7 && tid < 32) {
            tdm_load_2d((unsigned int)(uintptr_t)&alds[(ks + 1) & 1][0],
                        W + (size_t)m0 * CC + (ks + 1) * 32, 32u, 128u,
                        (unsigned int)CC, TDM_BF16_PAD40);
        }
        // stage B^T: 128 pixels x 32 channels, convert to bf16
#pragma unroll
        for (int i = 0; i < 16; ++i) {
            int idx = tid + i * 256;
            int pp = idx & 127, kc = idx >> 7;
            float xv = (float)xb[(size_t)(k0 + kc) * HW + p0 + pp];
            blds[pp * 40 + kc] = (__bf16)xv;
        }
        if (ks < 7) {   // hint next K-slab toward the caches (global_prefetch_b8)
            __builtin_prefetch(&xb[(size_t)(k0 + 32 + (tid >> 7)) * HW + p0 + (tid & 127)], 0, 1);
        }
        // allow the next tile's TDM to remain in flight (completes in order)
        if (tid < 32) {
            if (ks < 7) __builtin_amdgcn_s_wait_tensorcnt(1);
            else        __builtin_amdgcn_s_wait_tensorcnt(0);
        }
        __syncthreads();

        const __bf16* abuf = &alds[ks & 1][0];
        FragBF af[4], bf[2];
#pragma unroll
        for (int mi = 0; mi < 4; ++mi)
            af[mi].v = load_fragA(&abuf[(wm * 64 + mi * 16 + lm) * 40], lane);
#pragma unroll
        for (int ni = 0; ni < 2; ++ni)
            bf[ni].v = load_fragB(&blds[(wn * 32 + ni * 16 + lm) * 40], lane);
#pragma unroll
        for (int mi = 0; mi < 4; ++mi)
#pragma unroll
            for (int ni = 0; ni < 2; ++ni)
                acc[mi][ni] = WMMA_BF16(af[mi].v, bf[ni].v, acc[mi][ni]);
        __syncthreads();
    }

    int hi = (lane < 16) ? 0 : 8;
#pragma unroll
    for (int mi = 0; mi < 4; ++mi)
#pragma unroll
        for (int ni = 0; ni < 2; ++ni) {
            int colg = p0 + wn * 32 + ni * 16 + lm;
#pragma unroll
            for (int r = 0; r < 8; ++r) {
                int o = m0 + wm * 64 + mi * 16 + r + hi;
                float v = acc[mi][ni][r] + bias[o];
                Y[((size_t)b * MTOT + o) * HW + colg] = (TOUT)v;
            }
        }
}

// ---------------------------------------------------------------------------
// Attention: one block per (batch, window, head); 128 threads = 4 waves.
//   S = Q^T K * scale + bias + mask ; P = softmax(S) ; O = P V^T
//   Shift folded into addresses (uniform per head: hd>=4 -> +4, wraps mod 64,
//   so q/k/v gathers stay on the manual path; bias table comes in via TDM).
// ---------------------------------------------------------------------------
__global__ void __launch_bounds__(128)
attn_kernel(const __bf16* __restrict__ QKV, const float* __restrict__ biasTab,
            __bf16* __restrict__ AO) {
    int bi  = blockIdx.x;
    int b   = bi >> 9;            // 64 windows * 8 heads = 512 per batch
    int rem = bi & 511;
    int nw  = rem >> 3;
    int hd  = rem & 7;
    int wh  = nw >> 3, ww = nw & 7;
    int sh  = (hd >= 4) ? SHIFT_ : 0;

    __shared__ __bf16 qT[64 * 40];   // [t][c]
    __shared__ __bf16 kT[64 * 40];   // [s][c]
    __shared__ __bf16 vT[32 * 72];   // [c][s]
    __shared__ __bf16 P [64 * 72];   // [t][s]
    __shared__ float  bLDS[64 * 64]; // rel-pos bias for this head (TDM)

    int tid = threadIdx.x;

    // async: pull this head's 16KB bias table into LDS while we gather q/k/v
    if (tid < 32) {
        tdm_load_2d((unsigned int)(uintptr_t)&bLDS[0],
                    biasTab + (size_t)hd * 4096, 4096u, 1u, 4096u, TDM_F32_FLAT);
    }

    size_t base = (size_t)b * OQKV * HW;
    const __bf16* qb = QKV + base + (size_t)(hd * DK_) * HW;
    const __bf16* kb = QKV + base + (size_t)(CC + hd * DK_) * HW;
    const __bf16* vb = QKV + base + (size_t)(2 * CC + hd * DK_) * HW;

#pragma unroll
    for (int i = 0; i < 16; ++i) {
        int idx = tid + i * 128;
        int t = idx & 63, c = idx >> 6;
        int pi = (((wh * 8 + (t >> 3) + sh) & 63) << 6) | ((ww * 8 + (t & 7) + sh) & 63);
        qT[t * 40 + c] = qb[(size_t)c * HW + pi];
        kT[t * 40 + c] = kb[(size_t)c * HW + pi];
    }
#pragma unroll
    for (int i = 0; i < 16; ++i) {
        int idx = tid + i * 128;
        int s = idx & 63, c = idx >> 6;
        int pi = (((wh * 8 + (s >> 3) + sh) & 63) << 6) | ((ww * 8 + (s & 7) + sh) & 63);
        vT[c * 72 + s] = vb[(size_t)c * HW + pi];
    }
    if (tid < 32) __builtin_amdgcn_s_wait_tensorcnt(0);
    __syncthreads();

    int lane = tid & 31;
    int w    = tid >> 5;           // wave handles t rows [16w, 16w+16)
    int lm   = lane & 15;
    int hi   = (lane < 16) ? 0 : 8;

    // ---- S = Q^T K : M=16 (t), N=64 (s), K=32 ----
    v8f z = {0.f, 0.f, 0.f, 0.f, 0.f, 0.f, 0.f, 0.f};
    v16bf aq = load_fragA(&qT[(w * 16 + lm) * 40], lane);
    v8f acc[4];
#pragma unroll
    for (int nb = 0; nb < 4; ++nb) {
        v16bf bk = load_fragB(&kT[(nb * 16 + lm) * 40], lane);
        acc[nb] = WMMA_BF16(aq, bk, z);
    }

    // ---- scale + rel-pos bias + shifted-window mask (r outer: t-region hoisted) ----
    const float scale = 0.17677669529663687f;   // 1/sqrt(32)
    int s_col = lm;
#pragma unroll
    for (int r = 0; r < 8; ++r) {
        int t  = w * 16 + r + hi;
        int ht = wh * 8 + (t >> 3), wt = ww * 8 + (t & 7);
        int rt = (ht < 56 ? 0 : (ht < 60 ? 1 : 2)) * 3 + (wt < 56 ? 0 : (wt < 60 ? 1 : 2));
#pragma unroll
        for (int nb = 0; nb < 4; ++nb) {
            int s = nb * 16 + s_col;
            float v = acc[nb][r] * scale + bLDS[t * 64 + s];
            int hs = wh * 8 + (s >> 3), wsx = ww * 8 + (s & 7);
            int rs = (hs < 56 ? 0 : (hs < 60 ? 1 : 2)) * 3 + (wsx < 56 ? 0 : (wsx < 60 ? 1 : 2));
            if (rt != rs) v += -100.0f;
            acc[nb][r] = v;
        }
    }

    // ---- row softmax over s; butterfly via ds_swizzle (stays in 16-lane half) ----
#pragma unroll
    for (int r = 0; r < 8; ++r) {
        float m = fmaxf(fmaxf(acc[0][r], acc[1][r]), fmaxf(acc[2][r], acc[3][r]));
        m = fmaxf(m, swz_xor<1>(m));
        m = fmaxf(m, swz_xor<2>(m));
        m = fmaxf(m, swz_xor<4>(m));
        m = fmaxf(m, swz_xor<8>(m));
        float sum = 0.f;
#pragma unroll
        for (int nb = 0; nb < 4; ++nb) { float e = __expf(acc[nb][r] - m); acc[nb][r] = e; sum += e; }
        sum += swz_xor<1>(sum);
        sum += swz_xor<2>(sum);
        sum += swz_xor<4>(sum);
        sum += swz_xor<8>(sum);
        float inv = 1.0f / sum;
        int t = w * 16 + r + hi;
#pragma unroll
        for (int nb = 0; nb < 4; ++nb)
            P[t * 72 + nb * 16 + lm] = (__bf16)(acc[nb][r] * inv);
    }
    __syncthreads();

    // ---- O = P * V^T : M=16 (t), N=32 (c), K=64 ----
    v16bf pa[2];
#pragma unroll
    for (int kk = 0; kk < 2; ++kk)
        pa[kk] = load_fragA(&P[(w * 16 + lm) * 72 + kk * 32], lane);

    v8f oacc[2];
#pragma unroll
    for (int nb = 0; nb < 2; ++nb) {
        v8f o = z;
#pragma unroll
        for (int kk = 0; kk < 2; ++kk) {
            v16bf bv = load_fragB(&vT[(nb * 16 + lm) * 72 + kk * 32], lane);
            o = WMMA_BF16(pa[kk], bv, o);
        }
        oacc[nb] = o;
    }

#pragma unroll
    for (int nb = 0; nb < 2; ++nb)
#pragma unroll
        for (int r = 0; r < 8; ++r) {
            int t  = w * 16 + r + hi;
            int cg = hd * DK_ + nb * 16 + lm;
            int pi = (((wh * 8 + (t >> 3) + sh) & 63) << 6) | ((ww * 8 + (t & 7) + sh) & 63);
            AO[((size_t)b * CC + cg) * HW + pi] = (__bf16)oacc[nb][r];
        }
}

// ---------------------------------------------------------------------------
// Launch
// ---------------------------------------------------------------------------
extern "C" void kernel_launch(void* const* d_in, const int* in_sizes, int n_in,
                              void* d_out, int out_size, void* d_ws, size_t ws_size,
                              hipStream_t stream) {
    const float* x     = (const float*)d_in[0];
    const float* wq    = (const float*)d_in[1];
    const float* bq    = (const float*)d_in[2];
    const float* wkv   = (const float*)d_in[3];
    const float* bkv   = (const float*)d_in[4];
    const float* wproj = (const float*)d_in[5];
    const float* bproj = (const float*)d_in[6];
    const float* rpb   = (const float*)d_in[7];

    char* ws = (char*)d_ws;
    __bf16* Wall    = (__bf16*)(ws + 0);                       // 768*256*2   = 384 KB
    __bf16* Wproj   = (__bf16*)(ws + 393216);                  // 256*256*2   = 128 KB
    float*  ball    = (float*) (ws + 524288);                  // 768*4
    float*  biasTab = (float*) (ws + 528384);                  // 8*64*64*4   = 128 KB
    __bf16* QKV     = (__bf16*)(ws + 1048576);                 // 32*768*4096*2 = 192 MB
    __bf16* AO      = (__bf16*)(ws + 1048576 + 201326592);     // 32*256*4096*2 = 64 MB

    prep_kernel<<<256, 256, 0, stream>>>(wq, bq, wkv, bkv, wproj, rpb,
                                         Wall, Wproj, ball, biasTab);
    gemm_kernel<float, __bf16, OQKV><<<32 * 6 * 32, 256, 0, stream>>>(x, Wall, ball, QKV);
    attn_kernel<<<BB * 64 * HEADS_, 128, 0, stream>>>(QKV, biasTab, AO);
    gemm_kernel<__bf16, float, CC><<<32 * 2 * 32, 256, 0, stream>>>(AO, Wproj, bproj, (float*)d_out);
}
